// Attention_15092515078765
// MI455X (gfx1250) — matrix operand order repaired and checked by reading, not compiled, not run on hardware
//
#include <hip/hip_runtime.h>
#include <hip/hip_bf16.h>

typedef __attribute__((ext_vector_type(16))) _Float16 v16h;
typedef __attribute__((ext_vector_type(8)))  float    v8f;
typedef __attribute__((ext_vector_type(4)))  unsigned int v4u;
typedef __attribute__((ext_vector_type(4)))  int v4i;

union Frag16 { v16h h; v4u u[2]; };
union Win16  { v4u u[2]; _Float16 h[16]; };

#if defined(__gfx1250__) && \
    __has_builtin(__builtin_amdgcn_global_load_async_to_lds_b128) && \
    __has_builtin(__builtin_amdgcn_s_wait_asynccnt)
#define HAVE_ASYNC_LDS 1
#else
#define HAVE_ASYNC_LDS 0
#endif

// ---------------------------------------------------------------------------
// zero fill (16B vector stores)
// ---------------------------------------------------------------------------
__global__ void zero_f16_kernel(_Float16* __restrict__ p, long n8) {
    long i = (long)blockIdx.x * 256 + threadIdx.x;
    if (i < n8) ((v4u*)p)[i] = (v4u){0u, 0u, 0u, 0u};
}

// ---------------------------------------------------------------------------
// x (f32 NCHW) -> f16 halo-padded layout
// ---------------------------------------------------------------------------
__global__ void pad_x_f16(const float* __restrict__ in, _Float16* __restrict__ out) {
    const long total = 32L * 3 * 160 * 96;
    long i = (long)blockIdx.x * 256 + threadIdx.x;
    if (i >= total) return;
    int x = (int)(i % 96); long r = i / 96;
    int y = (int)(r % 160); r /= 160;
    int ci = (int)(r % 3); int img = (int)(r / 3);
    out[(size_t)(img * 3 + ci) * 16848 + (size_t)(y + 1) * 104 + (x + 1)] =
        (_Float16)in[i];
}

// ---------------------------------------------------------------------------
// Pre-swizzle conv weights (OIHW f32) into WMMA A-fragment-ready f16 layout:
//   out[(((rs*nCC + cc)*mtiles + mt)*32 + lane)*16 + j]
// ---------------------------------------------------------------------------
__global__ void swizzle_weights(const float* __restrict__ w,
                                _Float16* __restrict__ out, int Cin, int Cout) {
    long n = (long)Cout * Cin * 9;
    long i = (long)blockIdx.x * blockDim.x + threadIdx.x;
    if (i >= n) return;
    int j    = (int)(i & 15);
    int lane = (int)((i >> 4) & 31);
    long rest = i >> 9;
    int mtiles = Cout >> 4, nCC = Cin >> 5;
    int mt = (int)(rest % mtiles); rest /= mtiles;
    int cc = (int)(rest % nCC);
    int rs = (int)(rest / nCC);
    int mrow = lane & 15, hi = lane >> 4;
    int K  = j + hi * 8 + ((j >= 8) ? 8 : 0);
    int m  = mt * 16 + mrow;
    int ci = cc * 32 + K;
    out[i] = (_Float16)w[((long)m * Cin + ci) * 9 + rs];
}

// ---------------------------------------------------------------------------
// First layer (Cin=3): direct conv on padded layout, no bounds checks.
// ---------------------------------------------------------------------------
__global__ __launch_bounds__(256) void conv_first_f16(
    const _Float16* __restrict__ in, const float* __restrict__ w,
    const float* __restrict__ bias, _Float16* __restrict__ out)
{
    const long total = 32L * 64 * 160 * 96;
    long i = (long)blockIdx.x * 256 + threadIdx.x;
    if (i >= total) return;
    int x = (int)(i % 96); long r1 = i / 96;
    int y = (int)(r1 % 160); r1 /= 160;
    int co = (int)(r1 % 64); int img = (int)(r1 / 64);
    float s = bias[co];
    #pragma unroll
    for (int ci = 0; ci < 3; ++ci) {
        const _Float16* pl = in + (size_t)(img * 3 + ci) * 16848;
        #pragma unroll
        for (int dy = 0; dy < 3; ++dy)
            #pragma unroll
            for (int dx = 0; dx < 3; ++dx)
                s += (float)pl[(size_t)(y + dy) * 104 + (x + dx)] *
                     w[((co * 3 + ci) * 3 + dy) * 3 + dx];
    }
    out[(size_t)(img * 64 + co) * 16848 + (size_t)(y + 1) * 104 + (x + 1)] =
        (_Float16)fmaxf(s, 0.f);
}

// ---------------------------------------------------------------------------
// Implicit-GEMM 3x3 conv (pad 1) via WMMA on halo-padded activations.
// Cin%32==0, Cout%64==0. Loops: r(unrolled) -> cc -> s(unrolled).
// Per (r,cc): one ALIGNED 16-half window per segment (two b128 loads) covers
// all three s taps; tap extraction uses compile-time register indices.
// Block: 4 waves, macro-tile M=64 (4 WMMA m-tiles) x N=64 pixels ->
// 4 WMMAs per wave per K-chunk against one B fragment.
// ---------------------------------------------------------------------------
__global__ __launch_bounds__(128) void conv3x3_wmma(
    const _Float16* __restrict__ in, const _Float16* __restrict__ wsw,
    const float* __restrict__ bias, _Float16* __restrict__ out,
    int Cin, int Cout, int H, int W, int Wp, int PS)
{
    __shared__ _Float16 ldsA[4 * 512] __attribute__((aligned(16)));    // 4KB
    __shared__ _Float16 ldsB[4 * 64 * 8] __attribute__((aligned(16))); // 4KB

    const int t    = threadIdx.x;
    const int lane = t & 31;
    const int wv   = t >> 5;
    const int hi   = lane >> 4;
    const int ncol = wv * 16 + (lane & 15);
    const int HW   = H * W;
    const int pix0 = blockIdx.x * 64;
    const int mt0  = blockIdx.y * 4;       // four 16-wide m-tiles
    const int nCC  = Cin >> 5;
    const int mtiles = Cout >> 4;
    const long ccStride = 32L * PS;

    // ---- per-thread segment decode: invariant over all loops ----
    bool segFast[2];
    long segBaseA[2];          // aligned window base: ... + (y+1)*Wp + x0
    int  lbBase[2];
    int  fDelta[2][8];
    #pragma unroll
    for (int ss = 0; ss < 2; ++ss) {
        const int seg = t + ss * 128;
        const int kk = seg >> 3;
        const int n0 = (seg & 7) * 8;
        const int p0 = pix0 + n0;
        const int img = p0 / HW, rem = p0 % HW;
        const int y = rem / W, x0 = rem % W;
        segFast[ss] = ((x0 + 7) < W) && ((x0 & 7) == 0);
        segBaseA[ss] = ((long)img * Cin + kk) * PS + (long)(y + 1) * Wp + x0;
        lbBase[ss]  = ((kk >> 3) * 64 + n0) * 8 + (kk & 7);
        #pragma unroll
        for (int p = 0; p < 8; ++p) {
            int pp = p0 + p;
            int img2 = pp / HW, rem2 = pp % HW;
            int y2 = rem2 / W, x2 = rem2 % W;
            fDelta[ss][p] = (img2 - img) * (Cin * PS) + (y2 - y) * Wp + (x2 - x0);
        }
    }

    v8f acc[4] = {};

    #pragma unroll
    for (int r = -1; r <= 1; ++r) {
        long ccOff = 0;
        for (int cc = 0; cc < nCC; ++cc, ccOff += ccStride) {
            // ---- aligned input windows: [x0 .. x0+15] of row y+r, per seg ----
            Win16 win[2];
            #pragma unroll
            for (int ss = 0; ss < 2; ++ss) {
                if (segFast[ss]) {
                    const v4u* wp =
                        (const v4u*)(in + segBaseA[ss] + (long)r * Wp + ccOff);
                    win[ss].u[0] = wp[0];
                    win[ss].u[1] = wp[1];
                }
            }
            #pragma unroll
            for (int s = -1; s <= 1; ++s) {
                const int rs = (r + 1) * 3 + (s + 1);
                // ---- A tile: 2048 halfs (4KB), contiguous, frag-ready ----
                const _Float16* ga =
                    wsw + ((size_t)(rs * nCC + cc) * mtiles + mt0) * 512;
#if HAVE_ASYNC_LDS
                __builtin_amdgcn_global_load_async_to_lds_b128(
                    (__attribute__((address_space(1))) v4i*)(ga + t * 8),
                    (__attribute__((address_space(3))) v4i*)(ldsA + t * 8), 0, 0);
                __builtin_amdgcn_global_load_async_to_lds_b128(
                    (__attribute__((address_space(1))) v4i*)(ga + 1024 + t * 8),
                    (__attribute__((address_space(3))) v4i*)(ldsA + 1024 + t * 8), 0, 0);
#else
                *(v4u*)(ldsA + t * 8) = *(const v4u*)(ga + t * 8);
                *(v4u*)(ldsA + 1024 + t * 8) = *(const v4u*)(ga + 1024 + t * 8);
#endif
                // ---- B tile: tap extraction from window (or slow gather) ----
                #pragma unroll
                for (int ss = 0; ss < 2; ++ss) {
                    const int lb = lbBase[ss];
                    if (segFast[ss]) {
                        #pragma unroll
                        for (int p = 0; p < 8; ++p)
                            ldsB[lb + p * 8] = win[ss].h[1 + s + p];
                    } else {
                        const long base =
                            segBaseA[ss] + (long)r * Wp + (s + 1) + ccOff;
                        #pragma unroll
                        for (int p = 0; p < 8; ++p)
                            ldsB[lb + p * 8] = in[base + fDelta[ss][p]];
                    }
                }
#if HAVE_ASYNC_LDS
                __builtin_amdgcn_s_wait_asynccnt(0);
#endif
                __syncthreads();

                // ---- fragments: 4 A m-tiles + 1 shared B ----
                Frag16 fa[4], fb;
                const v4u* ap = (const v4u*)ldsA;
                #pragma unroll
                for (int m = 0; m < 4; ++m) {
                    fa[m].u[0] = ap[m * 64 + lane * 2 + 0];
                    fa[m].u[1] = ap[m * 64 + lane * 2 + 1];
                }
                const v4u* bp = (const v4u*)ldsB;
                fb.u[0] = bp[(hi + 0) * 64 + ncol];
                fb.u[1] = bp[(hi + 2) * 64 + ncol];

                #pragma unroll
                for (int m = 0; m < 4; ++m)
                    acc[m] = __builtin_amdgcn_wmma_f32_16x16x32_f16(
                        false, fa[m].h, false, fb.h, (short)0, acc[m], false, false);
                __syncthreads();
            }
        }
    }

    // ---- epilogue: bias + relu, write padded f16 ----
    const int p   = pix0 + ncol;
    const int img = p / HW, rem = p % HW;
    const int y = rem / W, x = rem % W;
    const long obase = (long)img * Cout * PS + (long)(y + 1) * Wp + (x + 1);
    #pragma unroll
    for (int m = 0; m < 4; ++m) {
        #pragma unroll
        for (int i = 0; i < 8; ++i) {
            const int co = mt0 * 16 + m * 16 + i + hi * 8;
            float v = fmaxf(acc[m][i] + bias[co], 0.f);
            out[obase + (long)co * PS] = (_Float16)v;
        }
    }
}

// ---------------------------------------------------------------------------
// 2x2 max pool, padded f16 in -> padded f16 out
// ---------------------------------------------------------------------------
__global__ void maxpool2_f16(const _Float16* __restrict__ in,
                             _Float16* __restrict__ out,
                             int C, int Ho, int Wo,
                             int Wpi, int PSi, int Wpo, int PSo, long total) {
    long i = (long)blockIdx.x * blockDim.x + threadIdx.x;
    if (i >= total) return;
    int xo = (int)(i % Wo); long r = i / Wo;
    int yo = (int)(r % Ho); r /= Ho;
    int c  = (int)(r % C);  int img = (int)(r / C);
    const _Float16* p = in + (size_t)(img * C + c) * PSi +
                        (size_t)(2 * yo + 1) * Wpi + (2 * xo + 1);
    _Float16 a = p[0], b = p[1], cc = p[Wpi], d = p[Wpi + 1];
    _Float16 m0 = (a > b) ? a : b;
    _Float16 m1 = (cc > d) ? cc : d;
    out[(size_t)(img * C + c) * PSo + (size_t)(yo + 1) * Wpo + (xo + 1)] =
        (m0 > m1) ? m0 : m1;
}

// ---------------------------------------------------------------------------
// Final LN over 7680 per frame: gather from padded [512][7][8] -> dense f32
// ---------------------------------------------------------------------------
__global__ __launch_bounds__(256) void ln_final(
    const _Float16* __restrict__ src, const float* __restrict__ w,
    const float* __restrict__ b, float* __restrict__ out)
{
    __shared__ float red[256];
    int f = blockIdx.x, t = threadIdx.x;
    float s = 0.f;
    for (int d = t; d < 7680; d += 256) {
        int c = d / 15, rr = (d % 15) / 3, cx = d % 3;
        s += (float)src[(size_t)(f * 512 + c) * 56 + (size_t)(rr + 1) * 8 + (cx + 1)];
    }
    red[t] = s; __syncthreads();
    for (int st = 128; st > 0; st >>= 1) { if (t < st) red[t] += red[t + st]; __syncthreads(); }
    float mu = red[0] / 7680.f; __syncthreads();
    float v = 0.f;
    for (int d = t; d < 7680; d += 256) {
        int c = d / 15, rr = (d % 15) / 3, cx = d % 3;
        float e = (float)src[(size_t)(f * 512 + c) * 56 + (size_t)(rr + 1) * 8 + (cx + 1)] - mu;
        v += e * e;
    }
    red[t] = v; __syncthreads();
    for (int st = 128; st > 0; st >>= 1) { if (t < st) red[t] += red[t + st]; __syncthreads(); }
    float inv = rsqrtf(red[0] / 7680.f + 1e-5f);
    for (int d = t; d < 7680; d += 256) {
        int c = d / 15, rr = (d % 15) / 3, cx = d % 3;
        float e = (float)src[(size_t)(f * 512 + c) * 56 + (size_t)(rr + 1) * 8 + (cx + 1)];
        out[(long)f * 7680 + d] = (e - mu) * inv * w[d] + b[d];
    }
}

// ---------------------------------------------------------------------------
// Small conv: 512->16, 3x3, pad 0, 5x3 -> 3x1 per frame. f32 dense input.
// ---------------------------------------------------------------------------
__global__ __launch_bounds__(256) void cnn_conv_f32(
    const float* __restrict__ in, const float* __restrict__ w,
    const float* __restrict__ bias, float* __restrict__ out)
{
    __shared__ float red[256];
    int bid = blockIdx.x;
    int oy = bid % 3; int r2 = bid / 3;
    int co = r2 % 16; int f = r2 / 16;
    float s = 0.f;
    for (int i = threadIdx.x; i < 512 * 9; i += 256) {
        int ci = i / 9, rs = i % 9, r = rs / 3, sx = rs % 3;
        s += in[(long)f * 7680 + (ci * 5 + (oy + r)) * 3 + sx] * w[(co * 512 + ci) * 9 + rs];
    }
    red[threadIdx.x] = s; __syncthreads();
    for (int st = 128; st > 0; st >>= 1) { if (threadIdx.x < st) red[threadIdx.x] += red[threadIdx.x + st]; __syncthreads(); }
    if (threadIdx.x == 0) out[(f * 16 + co) * 3 + oy] = red[0] + bias[co];
}

// ---------------------------------------------------------------------------
// LeakyReLU + LayerNorm over 48 per frame, f32
// ---------------------------------------------------------------------------
__global__ __launch_bounds__(64) void leaky_ln48(
    const float* __restrict__ in, const float* __restrict__ w,
    const float* __restrict__ b, float* __restrict__ feats)
{
    __shared__ float buf[48];
    __shared__ float red[64];
    int f = blockIdx.x, t = threadIdx.x;
    float v = 0.f;
    if (t < 48) { v = in[f * 48 + t]; v = (v >= 0.f) ? v : 0.01f * v; buf[t] = v; }
    red[t] = (t < 48) ? v : 0.f; __syncthreads();
    for (int st = 32; st > 0; st >>= 1) { if (t < st) red[t] += red[t + st]; __syncthreads(); }
    float mu = red[0] * (1.f / 48.f); __syncthreads();
    float d = (t < 48) ? (buf[t] - mu) : 0.f;
    red[t] = d * d; __syncthreads();
    for (int st = 32; st > 0; st >>= 1) { if (t < st) red[t] += red[t + st]; __syncthreads(); }
    float inv = rsqrtf(red[0] * (1.f / 48.f) + 1e-5f);
    if (t < 48) feats[f * 48 + t] = (buf[t] - mu) * inv * w[t] + b[t];
}

// ---------------------------------------------------------------------------
// Sequential attention + 2-layer GRU scan + head. Single block, all in LDS.
// out: [0..10] logits (post LN3), [11..58] h0, [59..106] h1
// ---------------------------------------------------------------------------
__global__ __launch_bounds__(192) void rnn_head_kernel(
    const float* __restrict__ feats,
    const float* __restrict__ att_w, const float* __restrict__ att_b,
    const float* __restrict__ wih0, const float* __restrict__ whh0,
    const float* __restrict__ bih0, const float* __restrict__ bhh0,
    const float* __restrict__ wih1, const float* __restrict__ whh1,
    const float* __restrict__ bih1, const float* __restrict__ bhh1,
    const float* __restrict__ prev_state,
    const float* __restrict__ pre_w, const float* __restrict__ pre_b,
    const float* __restrict__ n2w, const float* __restrict__ n2b,
    const float* __restrict__ cls_w, const float* __restrict__ cls_b,
    const float* __restrict__ n3w, const float* __restrict__ n3b,
    float* __restrict__ out)
{
    const int Ld = 32;
    __shared__ float f[32 * 48];
    __shared__ float h[2][48];
    __shared__ float outv[48];
    __shared__ float inp[48];
    __shared__ float gi[144], gh[144];
    __shared__ float a[32];
    __shared__ float red[64];
    __shared__ float sc[2];
    __shared__ float o16[16], o11[11];

    int t = threadIdx.x;
    for (int i = t; i < 32 * 48; i += 192) f[i] = feats[i];
    if (t < 48) { h[0][t] = prev_state[t]; h[1][t] = prev_state[48 + t]; outv[t] = 0.f; }
    __syncthreads();

    for (int step = 0; step < Ld; ++step) {
        if (t < 64) red[t] = (t < 48) ? outv[t] * att_w[t] : 0.f;
        __syncthreads();
        for (int st = 32; st > 0; st >>= 1) { if (t < st) red[t] += red[t + st]; __syncthreads(); }
        if (t == 0) sc[0] = red[0] + att_b[0];
        __syncthreads();
        if (t < Ld) {
            float s = sc[0];
            for (int d = 0; d < 48; ++d) s += f[t * 48 + d] * att_w[48 + d];
            a[t] = s;
        }
        __syncthreads();
        if (t == 0) {
            float mx = a[0];
            for (int j = 1; j < Ld; ++j) mx = fmaxf(mx, a[j]);
            float sum = 0.f;
            for (int j = 0; j < Ld; ++j) { a[j] = expf(a[j] - mx); sum += a[j]; }
            sc[1] = 1.f / sum;
        }
        __syncthreads();
        if (t < 48) {
            float s = 0.f;
            for (int j = 0; j < Ld; ++j) s += a[j] * f[j * 48 + t];
            inp[t] = s * sc[1];
        }
        __syncthreads();
        for (int li = 0; li < 2; ++li) {
            const float* wih = li ? wih1 : wih0;
            const float* whh = li ? whh1 : whh0;
            const float* bih = li ? bih1 : bih0;
            const float* bhh = li ? bhh1 : bhh0;
            if (t < 144) {
                float si = bih[t], sh = bhh[t];
                for (int d = 0; d < 48; ++d) {
                    si += wih[t * 48 + d] * inp[d];
                    sh += whh[t * 48 + d] * h[li][d];
                }
                gi[t] = si; gh[t] = sh;
            }
            __syncthreads();
            if (t < 48) {
                float r = 1.f / (1.f + expf(-(gi[t] + gh[t])));
                float z = 1.f / (1.f + expf(-(gi[48 + t] + gh[48 + t])));
                float n = tanhf(gi[96 + t] + r * gh[96 + t]);
                float hv = (1.f - z) * n + z * h[li][t];
                h[li][t] = hv;
                inp[t] = hv;
            }
            __syncthreads();
        }
        if (t < 48) outv[t] = inp[t];
        __syncthreads();
    }

    if (t < 16) {
        float s = pre_b[t];
        for (int d = 0; d < 48; ++d) s += pre_w[t * 48 + d] * inp[d];
        s = (s >= 0.f) ? s : 0.01f * s;
        o16[t] = s;
    }
    __syncthreads();
    if (t == 0) {
        float mu = 0.f; for (int i = 0; i < 16; ++i) mu += o16[i]; mu *= (1.f / 16.f);
        float var = 0.f; for (int i = 0; i < 16; ++i) { float d = o16[i] - mu; var += d * d; }
        float inv = rsqrtf(var * (1.f / 16.f) + 1e-5f);
        for (int i = 0; i < 16; ++i) o16[i] = (o16[i] - mu) * inv * n2w[i] + n2b[i];
    }
    __syncthreads();
    if (t < 11) {
        float s = cls_b[t];
        for (int d = 0; d < 16; ++d) s += cls_w[t * 16 + d] * o16[d];
        o11[t] = s;
    }
    __syncthreads();
    if (t == 0) {
        float mu = 0.f; for (int i = 0; i < 11; ++i) mu += o11[i]; mu *= (1.f / 11.f);
        float var = 0.f; for (int i = 0; i < 11; ++i) { float d = o11[i] - mu; var += d * d; }
        float inv = rsqrtf(var * (1.f / 11.f) + 1e-5f);
        for (int i = 0; i < 11; ++i) out[i] = (o11[i] - mu) * inv * n3w[i] + n3b[i];
    }
    if (t < 48) { out[11 + t] = h[0][t]; out[59 + t] = h[1][t]; }
}

// ---------------------------------------------------------------------------
// Host orchestration
// ---------------------------------------------------------------------------
extern "C" void kernel_launch(void* const* d_in, const int* in_sizes, int n_in,
                              void* d_out, int out_size, void* d_ws, size_t ws_size,
                              hipStream_t stream)
{
    (void)in_sizes; (void)n_in; (void)out_size; (void)ws_size;
    const float* x = (const float*)d_in[0];

    struct LayerDef { int cin, cout, pool; };
    const LayerDef L[13] = {
        {  3,  64, 0}, { 64,  64, 1},
        { 64, 128, 0}, {128, 128, 1},
        {128, 256, 0}, {256, 256, 0}, {256, 256, 1},
        {256, 512, 0}, {512, 512, 0}, {512, 512, 1},
        {512, 512, 0}, {512, 512, 0}, {512, 512, 1}};
    struct Geo { int H, W, Wp, PS; };
    const Geo G[6] = {
        {160, 96, 104, 16848}, {80, 48, 56, 4592}, {40, 24, 32, 1344},
        {20, 12, 16, 352}, {10, 6, 8, 96}, {5, 3, 8, 56}};

    char* ws = (char*)d_ws;
    size_t off = 0;
    auto alloc = [&](size_t bytes) { size_t o = off; off += (bytes + 255) & ~(size_t)255; return o; };
    const size_t MAXACT = (size_t)32 * 64 * 16848;          // largest padded act
    _Float16* bufA = (_Float16*)(ws + alloc(MAXACT * 2));
    _Float16* bufB = (_Float16*)(ws + alloc(MAXACT * 2));
    size_t wElems = 0;
    for (int i = 1; i < 13; ++i) wElems += (size_t)L[i].cin * L[i].cout * 9;
    _Float16* wsw = (_Float16*)(ws + alloc(wElems * 2));
    float* ln0buf = (float*)(ws + alloc((size_t)32 * 7680 * 4));
    float* cnnout = (float*)(ws + alloc((size_t)32 * 48 * 4));
    float* feats  = (float*)(ws + alloc((size_t)32 * 48 * 4));

    auto zero = [&](_Float16* p, long nElems) {
        long n8 = nElems / 8;
        zero_f16_kernel<<<(unsigned)((n8 + 255) / 256), 256, 0, stream>>>(p, n8);
    };

    // input -> padded f16
    {
        long nIn = 32L * 3 * G[0].PS;
        zero(bufA, nIn);
        long n = 32L * 3 * 160 * 96;
        pad_x_f16<<<(unsigned)((n + 255) / 256), 256, 0, stream>>>(x, bufA);
    }
    // swizzle weights for layers 1..12
    {
        size_t woff = 0;
        for (int i = 1; i < 13; ++i) {
            long n = (long)L[i].cin * L[i].cout * 9;
            swizzle_weights<<<(unsigned)((n + 255) / 256), 256, 0, stream>>>(
                (const float*)d_in[3 + 2 * i], wsw + woff, L[i].cin, L[i].cout);
            woff += (size_t)n;
        }
    }

    // VGG stack, ping-pong, halo-padded activations
    _Float16* src = bufA;
    _Float16* dst = bufB;
    size_t woff = 0;
    int stage = 0;
    for (int i = 0; i < 13; ++i) {
        const Geo g = G[stage];
        const int Cin = L[i].cin, Cout = L[i].cout;
        zero(dst, 32L * Cout * g.PS);
        if (i == 0) {
            long tot = 32L * 64 * 160 * 96;
            conv_first_f16<<<(unsigned)((tot + 255) / 256), 256, 0, stream>>>(
                src, (const float*)d_in[3], (const float*)d_in[4], dst);
        } else {
            long npix = 32L * g.H * g.W;
            dim3 grid((unsigned)(npix / 64), (unsigned)(Cout / 64));
            conv3x3_wmma<<<grid, 128, 0, stream>>>(src, wsw + woff,
                                                   (const float*)d_in[4 + 2 * i], dst,
                                                   Cin, Cout, g.H, g.W, g.Wp, g.PS);
            woff += (size_t)Cin * Cout * 9;
        }
        if (L[i].pool) {
            const Geo gn = G[stage + 1];
            zero(src, 32L * Cout * gn.PS);
            long tot = 32L * Cout * gn.H * gn.W;
            maxpool2_f16<<<(unsigned)((tot + 255) / 256), 256, 0, stream>>>(
                dst, src, Cout, gn.H, gn.W, g.Wp, g.PS, gn.Wp, gn.PS, tot);
            ++stage;
        } else {
            _Float16* tmp = src; src = dst; dst = tmp;
        }
    }
    // src: padded [32][512][7][8] f16  (5x3 interior)

    ln_final<<<32, 256, 0, stream>>>(src, (const float*)d_in[29],
                                     (const float*)d_in[30], ln0buf);
    cnn_conv_f32<<<32 * 16 * 3, 256, 0, stream>>>(ln0buf, (const float*)d_in[31],
                                                  (const float*)d_in[32], cnnout);
    leaky_ln48<<<32, 64, 0, stream>>>(cnnout, (const float*)d_in[33],
                                      (const float*)d_in[34], feats);
    rnn_head_kernel<<<1, 192, 0, stream>>>(
        feats,
        (const float*)d_in[35], (const float*)d_in[36],
        (const float*)d_in[37], (const float*)d_in[38],
        (const float*)d_in[39], (const float*)d_in[40],
        (const float*)d_in[41], (const float*)d_in[42],
        (const float*)d_in[43], (const float*)d_in[44],
        (const float*)d_in[2],
        (const float*)d_in[45], (const float*)d_in[46],
        (const float*)d_in[47], (const float*)d_in[48],
        (const float*)d_in[49], (const float*)d_in[50],
        (const float*)d_in[51], (const float*)d_in[52],
        (float*)d_out);
}
